// ARFlow_87900800680386
// MI455X (gfx1250) — matrix operand order repaired and checked by reading, not compiled, not run on hardware
//
#include <hip/hip_runtime.h>
#include <hip/hip_bf16.h>

// ---------------------------------------------------------------------------
// Fused causal conv stack (64->256->256->128, k=3) + affine-coupling epilogue.
// bf16 WMMA (v_wmma_f32_16x16x32_bf16) with f32 accumulation; all
// intermediates live in LDS (time-major [t][ch], padded stride for
// conflict-free ds_load_b128). TT=64 output timesteps per workgroup with a
// 16-row halo recomputed per layer; rows with global t<0 are forced to zero
// to reproduce the reference's zero padding between layers.
// ---------------------------------------------------------------------------

typedef __attribute__((ext_vector_type(16))) __bf16       v16bf;
typedef __attribute__((ext_vector_type(8)))  float        v8f;
typedef __attribute__((ext_vector_type(4)))  float        v4f;
typedef __attribute__((ext_vector_type(4)))  unsigned int v4u;

union BF16x16 { v4u q[2]; v16bf v; };
union AccU    { v4f f4[2]; v8f  v; };
union Pack8   { v4u q;     __bf16 h[8]; };

// Geometry
static constexpr int TT     = 64;           // output timesteps per WG
static constexpr int HALO   = 16;           // left halo rows
static constexpr int LT     = TT + HALO;    // 80 rows in every LDS buffer
static constexpr int XS_STR = 72;           // 64 ch + 8 pad (bf16)
static constexpr int H_STR  = 264;          // 256 ch + 8 pad (bf16)

// ws layout (bf16 elements): w_in [0,49152) | w_mid [49152,245760) | w_out [245760,344064)
static constexpr int WS_MID = 49152;
static constexpr int WS_OUT = 245760;
static constexpr int WS_TOT = 344064;

// ---- A fragment: W[h][tap][cin] row-major bf16 in global (per ISA 16-bit
// A 16x32 layout: lane<16 holds K {0..7,16..23}, lane>=16 holds {8..15,24..31})
static __device__ __forceinline__ v16bf load_A(const __bf16* __restrict__ w,
                                               int m_g, int tap, int cin,
                                               int kb, int hi) {
  const __bf16* p = w + ((m_g * 3 + tap) * cin + kb * 32 + hi * 8);
  BF16x16 r;
  r.q[0] = *(const v4u*)p;          // K = hi*8 + 0..7
  r.q[1] = *(const v4u*)(p + 16);   // K = 16 + hi*8 + 0..7
  return r.v;
}

// ---- B fragment from LDS activation buffer [t][ch] (lane<16: K {0..7,16..23},
// lane>=16: K {8..15,24..31}; N = lane%16 selects the time column)
static __device__ __forceinline__ v16bf load_B(const __bf16* buf, int row,
                                               int stride, int kb, int hi) {
  const __bf16* p = buf + row * stride + kb * 32 + hi * 8;
  BF16x16 r;
  r.q[0] = *(const v4u*)p;          // K = hi*8 + 0..7
  r.q[1] = *(const v4u*)(p + 16);   // K = 16 + hi*8 + 0..7
  return r.v;
}

static __device__ __forceinline__ void store8_bf16(__bf16* p, v8f a, bool zero) {
  Pack8 u;
#pragma unroll
  for (int i = 0; i < 8; ++i) u.h[i] = zero ? (__bf16)0.0f : (__bf16)a[i];
  *(v4u*)p = u.q;
}

// ---- One conv layer (output channels = 256, LDS->LDS). TOFF: input row for
// output row r, tap k is (r - TOFF + k). CIN in {64,256}.
template <int CIN, int TOFF>
static __device__ __forceinline__ void conv_layer(
    const __bf16* __restrict__ wgt, const float* __restrict__ bias,
    const __bf16* in_lds, int in_str, __bf16* out_lds,
    int wv, int lane, int t0) {
  const int n  = lane & 15;
  const int hi = lane >> 4;
  constexpr int KB  = CIN / 32;
  constexpr int NTT = LT / 16;  // 5 time tiles (halo recompute)

  for (int ct = wv; ct < 16; ct += 8) {
    const int cbase = ct * 16;
    AccU acc[NTT];
    const float* bp = bias + cbase + 8 * hi;
    const v4f blo = *(const v4f*)bp;
    const v4f bhi = *(const v4f*)(bp + 4);
#pragma unroll
    for (int tt = 0; tt < NTT; ++tt) { acc[tt].f4[0] = blo; acc[tt].f4[1] = bhi; }

#pragma unroll
    for (int kb = 0; kb < KB; ++kb) {
      const v16bf a0 = load_A(wgt, cbase + n, 0, CIN, kb, hi);
      const v16bf a1 = load_A(wgt, cbase + n, 1, CIN, kb, hi);
      const v16bf a2 = load_A(wgt, cbase + n, 2, CIN, kb, hi);
#pragma unroll
      for (int tt = 0; tt < NTT; ++tt) {
        const int rb = tt * 16 + n - TOFF;
        v16bf b0 = load_B(in_lds, rb + 0, in_str, kb, hi);
        acc[tt].v = __builtin_amdgcn_wmma_f32_16x16x32_bf16(
            false, a0, false, b0, (short)0, acc[tt].v, false, false);
        v16bf b1 = load_B(in_lds, rb + 1, in_str, kb, hi);
        acc[tt].v = __builtin_amdgcn_wmma_f32_16x16x32_bf16(
            false, a1, false, b1, (short)0, acc[tt].v, false, false);
        v16bf b2 = load_B(in_lds, rb + 2, in_str, kb, hi);
        acc[tt].v = __builtin_amdgcn_wmma_f32_16x16x32_bf16(
            false, a2, false, b2, (short)0, acc[tt].v, false, false);
      }
    }
#pragma unroll
    for (int tt = 0; tt < NTT; ++tt) {
      const int row  = tt * 16 + n;
      const bool z   = (t0 - HALO + row) < 0;  // zero padding between layers
      store8_bf16(out_lds + row * H_STR + cbase + 8 * hi, acc[tt].v, z);
    }
  }
}

__global__ __launch_bounds__(256) void arflow_main(
    const float* __restrict__ x, const float* __restrict__ b_in,
    const float* __restrict__ b_mid, const float* __restrict__ b_out,
    const float* __restrict__ alpha, const float* __restrict__ beta,
    const __bf16* __restrict__ ws, float* __restrict__ outz) {
  __shared__ __align__(16) __bf16 xs[LT * XS_STR];   //  11.25 KB
  __shared__ __align__(16) __bf16 h1[LT * H_STR];    //  41.25 KB
  __shared__ __align__(16) __bf16 h2[LT * H_STR];    //  41.25 KB

  const int t0  = blockIdx.x * TT;
  const int b   = blockIdx.y;
  const int tid = threadIdx.x;

  // ---- stage x tile (time-major, bf16), zero for t<0; r-fastest => coalesced
  for (int idx = tid; idx < LT * 64; idx += 256) {
    const int r = idx % LT;
    const int c = idx / LT;
    const int t = t0 - HALO + r;
    float v = 0.0f;
    if (t >= 0) v = x[((size_t)b * 64 + c) * 8192 + t];
    xs[r * XS_STR + c] = (__bf16)v;
  }
  __syncthreads();

  const int wv   = tid >> 5;
  const int lane = tid & 31;

  conv_layer<64, 3>(ws, b_in, xs, XS_STR, h1, wv, lane, t0);   // taps x[t-3..t-1]
  __syncthreads();
  conv_layer<256, 2>(ws + WS_MID, b_mid, h1, H_STR, h2, wv, lane, t0);
  __syncthreads();

  // ---- layer 3 (256 -> 128) + coupling epilogue; only rows t >= t0
  {
    const int n  = lane & 15;
    const int hi = lane >> 4;
    const __bf16* wgt = ws + WS_OUT;
    for (int j = wv; j < 16; j += 8) {          // 4 ch-pairs x 4 time tiles
      const int ct = j & 3;
      const int tt = j >> 2;
      const int cs = ct * 16;                    // log_s channels
      const int cT = 64 + ct * 16;               // t-shift channels
      AccU accS, accT;
      {
        const float* ps = b_out + cs + 8 * hi;
        const float* pt = b_out + cT + 8 * hi;
        accS.f4[0] = *(const v4f*)ps; accS.f4[1] = *(const v4f*)(ps + 4);
        accT.f4[0] = *(const v4f*)pt; accT.f4[1] = *(const v4f*)(pt + 4);
      }
      const int rb = HALO + tt * 16 + n - 2;     // input row base (tap 0)
#pragma unroll
      for (int kb = 0; kb < 8; ++kb) {
        const v16bf as0 = load_A(wgt, cs + n, 0, 256, kb, hi);
        const v16bf as1 = load_A(wgt, cs + n, 1, 256, kb, hi);
        const v16bf as2 = load_A(wgt, cs + n, 2, 256, kb, hi);
        const v16bf at0 = load_A(wgt, cT + n, 0, 256, kb, hi);
        const v16bf at1 = load_A(wgt, cT + n, 1, 256, kb, hi);
        const v16bf at2 = load_A(wgt, cT + n, 2, 256, kb, hi);
        v16bf b0 = load_B(h2, rb + 0, H_STR, kb, hi);
        accS.v = __builtin_amdgcn_wmma_f32_16x16x32_bf16(false, as0, false, b0, (short)0, accS.v, false, false);
        accT.v = __builtin_amdgcn_wmma_f32_16x16x32_bf16(false, at0, false, b0, (short)0, accT.v, false, false);
        v16bf b1 = load_B(h2, rb + 1, H_STR, kb, hi);
        accS.v = __builtin_amdgcn_wmma_f32_16x16x32_bf16(false, as1, false, b1, (short)0, accS.v, false, false);
        accT.v = __builtin_amdgcn_wmma_f32_16x16x32_bf16(false, at1, false, b1, (short)0, accT.v, false, false);
        v16bf b2 = load_B(h2, rb + 2, H_STR, kb, hi);
        accS.v = __builtin_amdgcn_wmma_f32_16x16x32_bf16(false, as2, false, b2, (short)0, accS.v, false, false);
        accT.v = __builtin_amdgcn_wmma_f32_16x16x32_bf16(false, at2, false, b2, (short)0, accT.v, false, false);
      }
      const int t = t0 + tt * 16 + n;
#pragma unroll
      for (int i = 0; i < 8; ++i) {
        const int c = cs + 8 * hi + i;
        const float s  = accS.v[i];
        const float tr = accT.v[i];
        const float al = alpha[c];
        const float be = beta[c];
        const float xv = x[((size_t)b * 64 + c) * 8192 + t];
        outz[((size_t)b * 64 + c) * 8192 + t] =
            __expf(al * tanhf(s) + be) * xv + tr;
      }
    }
  }
}

// ---- prolog: convert/relayout weights f32 -> bf16 as W[h][tap][cin] in ws
__global__ __launch_bounds__(256) void arflow_prep(
    const float* __restrict__ w_in, const float* __restrict__ w_mid,
    const float* __restrict__ w_out, __bf16* __restrict__ ws) {
  const int i = blockIdx.x * 256 + threadIdx.x;
  if (i < WS_MID) {                         // w_in (256,64,3)
    const int h = i / 192, rem = i % 192, tap = rem >> 6, c = rem & 63;
    ws[i] = (__bf16)w_in[(h * 64 + c) * 3 + tap];
  } else if (i < WS_OUT) {                  // w_mid (256,256,3)
    const int j = i - WS_MID;
    const int h = j / 768, rem = j % 768, tap = rem >> 8, c = rem & 255;
    ws[i] = (__bf16)w_mid[(h * 256 + c) * 3 + tap];
  } else if (i < WS_TOT) {                  // w_out (128,256,3)
    const int j = i - WS_OUT;
    const int o = j / 768, rem = j % 768, tap = rem >> 8, c = rem & 255;
    ws[i] = (__bf16)w_out[(o * 256 + c) * 3 + tap];
  }
}

extern "C" void kernel_launch(void* const* d_in, const int* in_sizes, int n_in,
                              void* d_out, int out_size, void* d_ws, size_t ws_size,
                              hipStream_t stream) {
  const float* x     = (const float*)d_in[0];
  const float* w_in  = (const float*)d_in[1];
  const float* b_in  = (const float*)d_in[2];
  const float* w_mid = (const float*)d_in[3];
  const float* b_mid = (const float*)d_in[4];
  const float* w_out = (const float*)d_in[5];
  const float* b_out = (const float*)d_in[6];
  const float* alpha = (const float*)d_in[7];
  const float* beta  = (const float*)d_in[8];
  __bf16* ws  = (__bf16*)d_ws;
  float* outz = (float*)d_out;
  (void)in_sizes; (void)n_in; (void)out_size; (void)ws_size;

  arflow_prep<<<WS_TOT / 256, 256, 0, stream>>>(w_in, w_mid, w_out, ws);
  dim3 grid(8192 / TT, 32);                 // (time tiles, batch)
  arflow_main<<<grid, 256, 0, stream>>>(x, b_in, b_mid, b_out, alpha, beta, ws, outz);
}
